// theModel_14843406975621
// MI455X (gfx1250) — matrix operand order repaired and checked by reading, not compiled, not run on hardware
//
#include <hip/hip_runtime.h>

// ---------------- model dimensions ----------------
constexpr int Bc = 2, Sc = 4096, Dc = 768, Hc = 12, DHc = 64, Wc = 256, Lc = 4, Fc = 2048;
constexpr int DVc = 256, DIDc = 32, Vc = 4, CDc = 16, CSc = 64, DEMBc = 1024;
constexpr int KINc = DVc + DIDc;          // 288
constexpr int BSc = Bc * Sc;              // 8192
#define NEGV (-1.0e9f)

// ---------------- WMMA types (wave32, gfx1250) ----------------
typedef _Float16 half_t;
typedef __attribute__((ext_vector_type(16))) _Float16 v16h;
typedef __attribute__((ext_vector_type(8)))  _Float16 v8h;
typedef __attribute__((ext_vector_type(8)))  float    v8f;

__device__ __forceinline__ v8f wmma16(v16h a, v16h b, v8f c) {
  // D = A(16x32 f16) * B(32x16 f16) + C(16x16 f32)
  return __builtin_amdgcn_wmma_f32_16x16x32_f16(false, a, false, b, (short)0, c, false, false);
}

__device__ __forceinline__ v16h cat16(v8h lo, v8h hi) {
  v16h r;
#pragma unroll
  for (int i = 0; i < 8; ++i) { r[i] = lo[i]; r[i + 8] = hi[i]; }
  return r;
}

// ---------------- CDNA5 async LDS DMA (ASYNCcnt-tracked) ----------------
// GV form: per-lane LDS byte offset in vdst VGPR, 64-bit global address in VGPR pair.
__device__ __forceinline__ void async_ld_b128(unsigned lds_off, const half_t* g) {
  asm volatile("global_load_async_to_lds_b128 %0, %1, off"
               :: "v"(lds_off), "v"(g) : "memory");
}
__device__ __forceinline__ void wait_async0() {
  asm volatile("s_wait_asynccnt 0x0" ::: "memory");
}
__device__ __forceinline__ void wait_async3() {   // allow the 3 in-flight ops of next tile
  asm volatile("s_wait_asynccnt 0x3" ::: "memory");
}

// =====================================================================
// Double-buffered WMMA GEMM:  C[M,N] = epi(A[M,K] @ W[K,N] + bias)
//   A:  f16 row-major (M x K)         (activations, produced in f16)
//   Bt: f16 pre-transposed (N x K)    (weights converted once per launch)
// Block = 256 thr (8 waves), tile 128x64, K-step 32; waves 4(M) x 2(N),
// each wave owns 32x32 (4 WMMAs / round). LDS tiles filled by
// global_load_async_to_lds_b128; tile i+1 DMA overlaps tile i WMMAs,
// synced with s_wait_asynccnt<=3 (each wave issues exactly 3 ops/tile;
// async loads complete in order). OOB rows/cols are clamped: duplicated
// data only feeds C rows/cols that the epilogue masks out.
// =====================================================================
enum { EPI_NONE = 0, EPI_SILU = 1, EPI_MUL = 2, EPI_ADDTO = 3 };
constexpr int BMg = 128, BNg = 64, BKg = 32;
constexpr int LDAg = 40;   // halfs/row (80B, multiple of 16B for v8h reads)
constexpr int LDBg = 48;   // halfs/row (96B, multiple of 32B for v16h reads)

template <int EPI, typename OutT>
__global__ __launch_bounds__(256)
void gemm_kernel(const half_t* __restrict__ A, const half_t* __restrict__ Bt,
                 const float* __restrict__ bias, const half_t* __restrict__ aux,
                 OutT* __restrict__ C, int M, int N, int K)
{
  __shared__ __align__(128) half_t Asub[2][BMg * LDAg];
  __shared__ __align__(128) half_t Bsub[2][BNg * LDBg];

  const int tid  = threadIdx.x;
  const int lane = tid & 31, wv = tid >> 5;
  const int hf   = lane >> 4, l15 = lane & 15;
  const int bm   = blockIdx.y * BMg, bn = blockIdx.x * BNg;
  const int wm   = (wv & 3) * 32,    wn = (wv >> 2) * 32;

  const unsigned abase0 = (unsigned)(size_t)&Asub[0][0];
  const unsigned abase1 = (unsigned)(size_t)&Asub[1][0];
  const unsigned bbase0 = (unsigned)(size_t)&Bsub[0][0];
  const unsigned bbase1 = (unsigned)(size_t)&Bsub[1][0];

  // staging coordinates: 256 lanes x b128; A needs 2 rounds (128 rows), B one (64 rows)
  const int srow = tid >> 2;             // 0..63
  const int sseg = (tid & 3) * 8;        // halfs 0,8,16,24 (16B chunks)
  // clamped source rows: always issue exactly 3 asyncs/tile per wave
  const int gr0 = (bm + srow      < M) ? (bm + srow)      : (M - 1);
  const int gr1 = (bm + srow + 64 < M) ? (bm + srow + 64) : (M - 1);
  const int gn  = (bn + srow      < N) ? (bn + srow)      : (N - 1);
  const unsigned aoff0 = (unsigned)((srow * LDAg + sseg) * 2);
  const unsigned aoff1 = (unsigned)(((srow + 64) * LDAg + sseg) * 2);
  const unsigned boff  = (unsigned)((srow * LDBg + sseg) * 2);

  v8f acc[2][2];
#pragma unroll
  for (int i = 0; i < 2; ++i)
#pragma unroll
    for (int j = 0; j < 2; ++j) acc[i][j] = {};

  // prologue: stage tile 0 into buffer 0
  async_ld_b128(abase0 + aoff0, A + (size_t)gr0 * K + sseg);
  async_ld_b128(abase0 + aoff1, A + (size_t)gr1 * K + sseg);
  async_ld_b128(bbase0 + boff,  Bt + (size_t)gn * K + sseg);

  int cur = 0;
  for (int k0 = 0; k0 < K; k0 += BKg) {
    const bool more = (k0 + BKg) < K;
    __syncthreads();                       // everyone done reading buf[cur^1]
    if (more) {                            // stage tile i+1 into the other buffer
      const int kn = k0 + BKg;
      const unsigned ab = cur ? abase0 : abase1;
      const unsigned bb = cur ? bbase0 : bbase1;
      async_ld_b128(ab + aoff0, A + (size_t)gr0 * K + kn + sseg);
      async_ld_b128(ab + aoff1, A + (size_t)gr1 * K + kn + sseg);
      async_ld_b128(bb + boff,  Bt + (size_t)gn * K + kn + sseg);
      wait_async3();                       // tile i resident (in-order completion)
    } else {
      wait_async0();
    }
    __syncthreads();                       // all waves' waits done -> buf[cur] valid

    const half_t* As = &Asub[cur][0];
    const half_t* Bs = &Bsub[cur][0];
    const int ka = hf * 16;
    v16h a0 = cat16(*(const v8h*)&As[(wm + l15) * LDAg + ka],
                    *(const v8h*)&As[(wm + l15) * LDAg + ka + 8]);
    v16h a1 = cat16(*(const v8h*)&As[(wm + 16 + l15) * LDAg + ka],
                    *(const v8h*)&As[(wm + 16 + l15) * LDAg + ka + 8]);
    v16h b0 = *(const v16h*)&Bs[(wn + l15) * LDBg + hf * 16];
    v16h b1 = *(const v16h*)&Bs[(wn + 16 + l15) * LDBg + hf * 16];

    acc[0][0] = wmma16(a0, b0, acc[0][0]);
    acc[0][1] = wmma16(a0, b1, acc[0][1]);
    acc[1][0] = wmma16(a1, b0, acc[1][0]);
    acc[1][1] = wmma16(a1, b1, acc[1][1]);
    cur ^= 1;
  }

  // epilogue: D layout — lane col = l15, vgpr v -> row v + 8*hf
#pragma unroll
  for (int jt = 0; jt < 2; ++jt) {
    const int col = bn + wn + jt * 16 + l15;
    if (col < N) {
      const float bv = bias[col];
#pragma unroll
      for (int it = 0; it < 2; ++it) {
#pragma unroll
        for (int vi = 0; vi < 8; ++vi) {
          const int row = bm + wm + it * 16 + vi + 8 * hf;
          if (row < M) {
            float val = acc[it][jt][vi] + bv;
            if (EPI == EPI_SILU) val = val / (1.f + __expf(-val));
            const size_t idx = (size_t)row * N + col;
            if (EPI == EPI_MUL) val *= (float)aux[idx];
            if (EPI == EPI_ADDTO) C[idx] = (OutT)((float)C[idx] + val);
            else                  C[idx] = (OutT)val;
          }
        }
      }
    }
  }
}

template <int EPI, typename OutT>
static void gemm_launch(const half_t* A, const half_t* Bt, const float* bias,
                        const half_t* aux, OutT* C, int M, int N, int K, hipStream_t s)
{
  dim3 grid((N + BNg - 1) / BNg, (M + BMg - 1) / BMg);
  gemm_kernel<EPI, OutT><<<grid, dim3(256), 0, s>>>(A, Bt, bias, aux, C, M, N, K);
}

// =====================================================================
// Sliding-window flash attention, one block per (chunk, head, batch).
// q,k,v are f16 (B,S,D), head hd at columns [hd*64, hd*64+64).  h += softmax(qk^T)v
// =====================================================================
__global__ __launch_bounds__(256)
void attn_kernel(const half_t* __restrict__ Q, const half_t* __restrict__ Kx,
                 const half_t* __restrict__ Vx, float* __restrict__ H)
{
  __shared__ __align__(128) half_t Ks[64 * 64];     // [key][dh]  -> B-operand for scores
  __shared__ __align__(128) half_t VTs[64 * 72];    // [dh][key]  -> B-operand for PV
  __shared__ __align__(128) half_t Ps[8 * 32 * 64]; // per-wave probs [32 q][64 key]

  const int c = blockIdx.x, hd = blockIdx.y, b = blockIdx.z;
  const int tid = threadIdx.x, lane = tid & 31, wv = tid >> 5;
  const int hf = lane >> 4, l15 = lane & 15;
  const int qbase = wv * 32;

  // Q fragments in registers, scaled by 1/sqrt(DH)
  v16h qf[2][2];
#pragma unroll
  for (int mt = 0; mt < 2; ++mt) {
    const int m = qbase + mt * 16 + l15;
    const half_t* qrow = Q + ((size_t)(b * Sc + c * Wc + m) * Dc) + hd * DHc;
#pragma unroll
    for (int ks = 0; ks < 2; ++ks) {
      v16h f;
#pragma unroll
      for (int e = 0; e < 16; ++e) {
        const int kd = ks * 32 + hf * 16 + e;
        f[e] = (half_t)((float)qrow[kd] * 0.125f);
      }
      qf[mt][ks] = f;
    }
  }

  v8f O[2][4];
#pragma unroll
  for (int mt = 0; mt < 2; ++mt)
#pragma unroll
    for (int dt = 0; dt < 4; ++dt) O[mt][dt] = {};
  float mrun[2][8], lrun[2][8];
#pragma unroll
  for (int mt = 0; mt < 2; ++mt)
#pragma unroll
    for (int vi = 0; vi < 8; ++vi) { mrun[mt][vi] = -1.0e30f; lrun[mt][vi] = 0.f; }

  const int kbmin = qbase >> 6;
  int kbmax = (qbase + 31 + 2 * Wc) >> 6; if (kbmax > 11) kbmax = 11;

  for (int kblk = 0; kblk < 12; ++kblk) {
    __syncthreads();
    { // stage 64-key block of K and V^T (zero-fill OOB so 0*Inf can't make NaNs)
      const int key = tid >> 2, dg = (tid & 3) * 16;
      const int kpos = kblk * 64 + key;
      const int s = c * Wc + kpos - Wc;
      if (s >= 0 && s < Sc) {
        const half_t* kr = Kx + ((size_t)(b * Sc + s) * Dc) + hd * DHc + dg;
        const half_t* vr = Vx + ((size_t)(b * Sc + s) * Dc) + hd * DHc + dg;
        *(v8h*)&Ks[key * 64 + dg]     = *(const v8h*)kr;
        *(v8h*)&Ks[key * 64 + dg + 8] = *(const v8h*)(kr + 8);
#pragma unroll
        for (int j = 0; j < 16; ++j) VTs[(dg + j) * 72 + key] = vr[j];
      } else {
#pragma unroll
        for (int j = 0; j < 16; ++j) {
          Ks[key * 64 + dg + j]    = (half_t)0.f;
          VTs[(dg + j) * 72 + key] = (half_t)0.f;
        }
      }
    }
    __syncthreads();
    if (kblk < kbmin || kblk > kbmax) continue;   // wave-uniform band skip

    // scores S = q @ k^T  (2 Mtiles x 4 Ntiles over 64 keys)
    v8f SD[2][4];
#pragma unroll
    for (int mt = 0; mt < 2; ++mt)
#pragma unroll
      for (int nt = 0; nt < 4; ++nt) SD[mt][nt] = {};
#pragma unroll
    for (int ks = 0; ks < 2; ++ks) {
#pragma unroll
      for (int nt = 0; nt < 4; ++nt) {
        v16h bf = *(const v16h*)&Ks[(nt * 16 + l15) * 64 + ks * 32 + hf * 16];
        SD[0][nt] = wmma16(qf[0][ks], bf, SD[0][nt]);
        SD[1][nt] = wmma16(qf[1][ks], bf, SD[1][nt]);
      }
    }

    // mask + online softmax (row reductions across half-wave)
#pragma unroll
    for (int mt = 0; mt < 2; ++mt) {
#pragma unroll
      for (int vi = 0; vi < 8; ++vi) {
        const int qpos = qbase + mt * 16 + vi + 8 * hf;
        float rmax = -3.0e38f;
#pragma unroll
        for (int nt = 0; nt < 4; ++nt) {
          const int kpos = kblk * 64 + nt * 16 + l15;
          const int sg = c * Wc + kpos - Wc;
          const bool valid = (kpos >= qpos) && (kpos <= qpos + 2 * Wc) && (sg >= 0) && (sg < Sc);
          const float sv = valid ? SD[mt][nt][vi] : NEGV;
          SD[mt][nt][vi] = sv;
          rmax = fmaxf(rmax, sv);
        }
#pragma unroll
        for (int d = 1; d <= 8; d <<= 1) rmax = fmaxf(rmax, __shfl_xor(rmax, d, 32));
        const float mold = mrun[mt][vi];
        const float mnew = fmaxf(mold, rmax);
        const float scal = __expf(mold - mnew);
        float rsum = 0.f;
#pragma unroll
        for (int nt = 0; nt < 4; ++nt) {
          const float p = __expf(SD[mt][nt][vi] - mnew);
          rsum += p;
          Ps[(wv * 32 + mt * 16 + vi + 8 * hf) * 64 + nt * 16 + l15] = (half_t)p;
        }
#pragma unroll
        for (int d = 1; d <= 8; d <<= 1) rsum += __shfl_xor(rsum, d, 32);
        lrun[mt][vi] = lrun[mt][vi] * scal + rsum;
        mrun[mt][vi] = mnew;
#pragma unroll
        for (int dt = 0; dt < 4; ++dt) O[mt][dt][vi] *= scal;
      }
    }

    // O += P @ V  (A-frags re-read from this wave's LDS region; same-wave DS is in-order)
#pragma unroll
    for (int mt = 0; mt < 2; ++mt) {
#pragma unroll
      for (int ks2 = 0; ks2 < 2; ++ks2) {
        const int prow = wv * 32 + mt * 16 + l15;
        const int pk0 = ks2 * 32 + hf * 16;
        v16h pa = cat16(*(const v8h*)&Ps[prow * 64 + pk0],
                        *(const v8h*)&Ps[prow * 64 + pk0 + 8]);
#pragma unroll
        for (int dt = 0; dt < 4; ++dt) {
          v16h bf = *(const v16h*)&VTs[(dt * 16 + l15) * 72 + ks2 * 32 + hf * 16];
          O[mt][dt] = wmma16(pa, bf, O[mt][dt]);
        }
      }
    }
  }

  // normalize + fused residual add into h (f32)
#pragma unroll
  for (int mt = 0; mt < 2; ++mt) {
#pragma unroll
    for (int vi = 0; vi < 8; ++vi) {
      const int qpos = qbase + mt * 16 + vi + 8 * hf;
      float li = lrun[mt][vi]; li = (li > 0.f) ? li : 1.f;
      const float inv = 1.f / li;
      const size_t base = ((size_t)(b * Sc + c * Wc + qpos) * Dc) + hd * DHc;
#pragma unroll
      for (int dt = 0; dt < 4; ++dt) {
        const size_t idx = base + dt * 16 + l15;
        H[idx] += O[mt][dt][vi] * inv;
      }
    }
  }
}

// =====================================================================
// small kernels
// =====================================================================

// weight convert + transpose: W (K x N) f32 -> Wt (N x K) f16, once per launch
__global__ __launch_bounds__(256)
void convt_kernel(const float* __restrict__ W, half_t* __restrict__ Wt, int K, int N)
{
  const size_t i = (size_t)blockIdx.x * 256 + threadIdx.x;
  if (i >= (size_t)K * N) return;
  const int n = (int)(i / (size_t)K);
  const int k = (int)(i - (size_t)n * K);
  Wt[i] = (half_t)W[(size_t)k * N + n];
}

__global__ __launch_bounds__(256)
void cvt16_kernel(const float* __restrict__ x, half_t* __restrict__ y, int n)
{
  const int i = blockIdx.x * 256 + threadIdx.x;
  if (i < n) y[i] = (half_t)x[i];
}

__global__ __launch_bounds__(256)
void embed_kernel(const int* __restrict__ x, const int* __restrict__ idv,
                  const float* __restrict__ table, half_t* __restrict__ xin)
{
  const int i = blockIdx.x * 256 + threadIdx.x;
  if (i >= BSc * KINc) return;
  const int row = i / KINc, col = i - row * KINc;
  float v;
  if (col < DVc) v = table[x[row] * DVc + col] * 16.0f;          // sqrt(DV)=16
  else           v = (float)((idv[row] >> (31 - (col - DVc))) & 1);
  xin[i] = (half_t)v;
}

template <typename OutT>
__global__ __launch_bounds__(256)
void rmsnorm_kernel(const float* __restrict__ x, const float* __restrict__ w,
                    OutT* __restrict__ y, int Dn)
{
  __shared__ float red[8];
  const int row = blockIdx.x;
  const float* xr = x + (size_t)row * Dn;
  float ss = 0.f;
  for (int i = threadIdx.x; i < Dn; i += 256) { const float t = xr[i]; ss += t * t; }
#pragma unroll
  for (int d = 16; d >= 1; d >>= 1) ss += __shfl_xor(ss, d, 32);
  if ((threadIdx.x & 31) == 0) red[threadIdx.x >> 5] = ss;
  __syncthreads();
  float tot = 0.f;
#pragma unroll
  for (int i = 0; i < 8; ++i) tot += red[i];
  const float rs = rsqrtf(tot / (float)Dn + 1e-6f);
  for (int i = threadIdx.x; i < Dn; i += 256) y[(size_t)row * Dn + i] = (OutT)(xr[i] * rs * w[i]);
}

__global__ __launch_bounds__(256)
void transpose_c_kernel(const half_t* __restrict__ csm, half_t* __restrict__ cT)
{
  const int i = blockIdx.x * 256 + threadIdx.x;   // over B*16*S
  if (i >= Bc * CDc * Sc) return;
  const int s = i & (Sc - 1);
  const int d = (i >> 12) & (CDc - 1);
  const int b = i >> 16;
  cT[i] = csm[((size_t)(b * Sc + s)) * CDc + d];
}

__global__ __launch_bounds__(256)
void embnorm_kernel(const float* __restrict__ cc, const float* __restrict__ w,
                    half_t* __restrict__ emb)
{
  __shared__ float buf[DEMBc];
  __shared__ float red[8];
  const int b = blockIdx.x;
  for (int i = threadIdx.x; i < DEMBc; i += 256) {
    const int cs = i >> 4, cd = i & 15;                 // emb[j] = c[b, cd, cs], j = cs*16+cd
    buf[i] = cc[(b * CDc + cd) * CSc + cs];
  }
  __syncthreads();
  float ss = 0.f;
  for (int i = threadIdx.x; i < DEMBc; i += 256) ss += buf[i] * buf[i];
#pragma unroll
  for (int d = 16; d >= 1; d >>= 1) ss += __shfl_xor(ss, d, 32);
  if ((threadIdx.x & 31) == 0) red[threadIdx.x >> 5] = ss;
  __syncthreads();
  float tot = 0.f;
#pragma unroll
  for (int i = 0; i < 8; ++i) tot += red[i];
  const float rs = rsqrtf(tot / (float)DEMBc + 1e-6f);
  for (int i = threadIdx.x; i < DEMBc; i += 256) emb[b * DEMBc + i] = (half_t)(buf[i] * rs * w[i]);
}

static void convt(const float* W, half_t* Wt, int K, int N, hipStream_t s)
{
  const size_t total = (size_t)K * N;
  convt_kernel<<<(unsigned)((total + 255) / 256), 256, 0, s>>>(W, Wt, K, N);
}

// =====================================================================
// launcher
// =====================================================================
extern "C" void kernel_launch(void* const* d_in, const int* in_sizes, int n_in,
                              void* d_out, int out_size, void* d_ws, size_t ws_size,
                              hipStream_t stream)
{
  const int*   x           = (const int*)  d_in[0];
  const int*   idv         = (const int*)  d_in[1];
  const float* embed_table = (const float*)d_in[2];
  const float* cd_w        = (const float*)d_in[3];
  const float* cd_b        = (const float*)d_in[4];
  const float* wq          = (const float*)d_in[5];
  const float* bq          = (const float*)d_in[6];
  const float* wk          = (const float*)d_in[7];
  const float* bk          = (const float*)d_in[8];
  const float* wv          = (const float*)d_in[9];
  const float* bv          = (const float*)d_in[10];
  const float* attn_norm_w = (const float*)d_in[11];
  const float* ffn_norm_w  = (const float*)d_in[12];
  const float* w1          = (const float*)d_in[13];
  const float* b1          = (const float*)d_in[14];
  const float* w2          = (const float*)d_in[15];
  const float* b2          = (const float*)d_in[16];
  const float* w3          = (const float*)d_in[17];
  const float* b3          = (const float*)d_in[18];
  const float* h0_w1       = (const float*)d_in[19];
  const float* h0_b1       = (const float*)d_in[20];
  const float* h0_w2       = (const float*)d_in[21];
  const float* h0_b2       = (const float*)d_in[22];
  const float* norm_head0_w= (const float*)d_in[23];
  const float* head0_w     = (const float*)d_in[24];
  const float* head0_b     = (const float*)d_in[25];
  const float* c1_w1       = (const float*)d_in[26];
  const float* c1_b1       = (const float*)d_in[27];
  const float* c1_w2       = (const float*)d_in[28];
  const float* c1_b2       = (const float*)d_in[29];
  const float* c2_w1       = (const float*)d_in[30];
  const float* c2_b1       = (const float*)d_in[31];
  const float* c2_w2       = (const float*)d_in[32];
  const float* c2_b2       = (const float*)d_in[33];
  const float* norm_emb_w  = (const float*)d_in[34];
  const float* head1_w     = (const float*)d_in[35];
  const float* head1_b     = (const float*)d_in[36];
  float* out = (float*)d_out;

  // ---- workspace carve (256B aligned bump allocator) ----
  char* p = (char*)d_ws;
  auto allocB = [&](size_t bytes) { char* r = p; p += (bytes + 255) & ~(size_t)255; return r; };
  const size_t MD = (size_t)BSc * Dc;
  const size_t MF = (size_t)BSc * Fc;

  float*  h    = (float*) allocB(MD * 4);
  float*  tf   = (float*) allocB(MD * 4);
  float*  cc   = (float*) allocB((size_t)Bc * CDc * CSc * 4);
  half_t* a16  = (half_t*)allocB(MD * 2);
  half_t* h16  = (half_t*)allocB(MD * 2);
  half_t* q16  = (half_t*)allocB(MD * 2);
  half_t* k16  = (half_t*)allocB(MD * 2);
  half_t* v16  = (half_t*)allocB(MD * 2);
  half_t* g1   = (half_t*)allocB(MF * 2);
  half_t* g3   = (half_t*)allocB(MF * 2);
  // f16 pre-transposed weights (NxK)
  half_t* cdwt   = (half_t*)allocB((size_t)KINc * Dc * 2);
  half_t* wqt    = (half_t*)allocB((size_t)Lc * Dc * Dc * 2);
  half_t* wkt    = (half_t*)allocB((size_t)Lc * Dc * Dc * 2);
  half_t* wvt    = (half_t*)allocB((size_t)Lc * Dc * Dc * 2);
  half_t* w1t    = (half_t*)allocB((size_t)Lc * Dc * Fc * 2);
  half_t* w3t    = (half_t*)allocB((size_t)Lc * Dc * Fc * 2);
  half_t* w2t    = (half_t*)allocB((size_t)Lc * Fc * Dc * 2);
  half_t* h0w1t  = (half_t*)allocB((size_t)Dc * Dc * 2);
  half_t* h0w2t  = (half_t*)allocB((size_t)Dc * Dc * 2);
  half_t* hd0wt  = (half_t*)allocB((size_t)Dc * Vc * 2);
  half_t* c1w1t  = (half_t*)allocB((size_t)Dc * Dc * 2);
  half_t* c1w2t  = (half_t*)allocB((size_t)Dc * CDc * 2);
  half_t* c2w1t  = (half_t*)allocB((size_t)Sc * Sc * 2);
  half_t* c2w2t  = (half_t*)allocB((size_t)Sc * CSc * 2);
  half_t* hd1wt  = (half_t*)allocB((size_t)DEMBc * Sc * Vc * 2);
  // aliases (regions free at time of use)
  half_t* xin  = g1;                               // pre-layer only
  half_t* cTh  = g3;                               // (32, 4096)
  half_t* ct2h = g3 + (size_t)Bc * CDc * Sc;       // (32, 4096)
  half_t* embv = ct2h + (size_t)Bc * CDc * Sc;     // (2, 1024)
  half_t* csm  = v16;                              // (B*S, 16)

  // ---- weight conversion (f32 KxN -> f16 NxK), once per launch ----
  convt(cd_w, cdwt, KINc, Dc, stream);
  for (int l = 0; l < Lc; ++l) {
    convt(wq + (size_t)l * Dc * Dc, wqt + (size_t)l * Dc * Dc, Dc, Dc, stream);
    convt(wk + (size_t)l * Dc * Dc, wkt + (size_t)l * Dc * Dc, Dc, Dc, stream);
    convt(wv + (size_t)l * Dc * Dc, wvt + (size_t)l * Dc * Dc, Dc, Dc, stream);
    convt(w1 + (size_t)l * Dc * Fc, w1t + (size_t)l * Dc * Fc, Dc, Fc, stream);
    convt(w3 + (size_t)l * Dc * Fc, w3t + (size_t)l * Dc * Fc, Dc, Fc, stream);
    convt(w2 + (size_t)l * Fc * Dc, w2t + (size_t)l * Fc * Dc, Fc, Dc, stream);
  }
  convt(h0_w1,  h0w1t, Dc, Dc, stream);
  convt(h0_w2,  h0w2t, Dc, Dc, stream);
  convt(head0_w, hd0wt, Dc, Vc, stream);
  convt(c1_w1,  c1w1t, Dc, Dc, stream);
  convt(c1_w2,  c1w2t, Dc, CDc, stream);
  convt(c2_w1,  c2w1t, Sc, Sc, stream);
  convt(c2_w2,  c2w2t, Sc, CSc, stream);
  convt(head1_w, hd1wt, DEMBc, Sc * Vc, stream);

  // ---- 1) embedding + id bits -> xin f16 (8192 x 288) -> h f32 (8192 x 768)
  embed_kernel<<<(BSc * KINc + 255) / 256, 256, 0, stream>>>(x, idv, embed_table, xin);
  gemm_launch<EPI_NONE, float>(xin, cdwt, cd_b, nullptr, h, BSc, Dc, KINc, stream);

  // ---- 2) transformer layers
  for (int l = 0; l < Lc; ++l) {
    rmsnorm_kernel<half_t><<<BSc, 256, 0, stream>>>(h, attn_norm_w + (size_t)l * Dc, a16, Dc);
    gemm_launch<EPI_NONE, half_t>(a16, wqt + (size_t)l * Dc * Dc, bq + (size_t)l * Dc, nullptr, q16, BSc, Dc, Dc, stream);
    gemm_launch<EPI_NONE, half_t>(a16, wkt + (size_t)l * Dc * Dc, bk + (size_t)l * Dc, nullptr, k16, BSc, Dc, Dc, stream);
    gemm_launch<EPI_NONE, half_t>(a16, wvt + (size_t)l * Dc * Dc, bv + (size_t)l * Dc, nullptr, v16, BSc, Dc, Dc, stream);
    attn_kernel<<<dim3(Sc / Wc, Hc, Bc), 256, 0, stream>>>(q16, k16, v16, h);   // h += attn
    rmsnorm_kernel<half_t><<<BSc, 256, 0, stream>>>(h, ffn_norm_w + (size_t)l * Dc, a16, Dc);
    gemm_launch<EPI_SILU,  half_t>(a16, w1t + (size_t)l * Dc * Fc, b1 + (size_t)l * Fc, nullptr, g1, BSc, Fc, Dc, stream);
    gemm_launch<EPI_MUL,   half_t>(a16, w3t + (size_t)l * Dc * Fc, b3 + (size_t)l * Fc, g1,      g3, BSc, Fc, Dc, stream);
    gemm_launch<EPI_ADDTO, float >(g3,  w2t + (size_t)l * Fc * Dc, b2 + (size_t)l * Dc, nullptr, h,  BSc, Dc, Fc, stream);
  }

  // ---- 3) head0
  cvt16_kernel<<<(int)((MD + 255) / 256), 256, 0, stream>>>(h, h16, (int)MD);
  gemm_launch<EPI_SILU, half_t>(h16, h0w1t, h0_b1, nullptr, q16, BSc, Dc, Dc, stream);
  gemm_launch<EPI_NONE, float >(q16, h0w2t, h0_b2, nullptr, tf,  BSc, Dc, Dc, stream);
  rmsnorm_kernel<half_t><<<BSc, 256, 0, stream>>>(tf, norm_head0_w, k16, Dc);
  gemm_launch<EPI_NONE, float >(k16, hd0wt, head0_b, nullptr, out, BSc, Vc, Dc, stream);

  // ---- 4) compression path -> logits
  gemm_launch<EPI_SILU, half_t>(h16, c1w1t, c1_b1, nullptr, q16, BSc, Dc, Dc, stream);
  gemm_launch<EPI_NONE, half_t>(q16, c1w2t, c1_b2, nullptr, csm, BSc, CDc, Dc, stream);
  transpose_c_kernel<<<(Bc * CDc * Sc + 255) / 256, 256, 0, stream>>>(csm, cTh);
  gemm_launch<EPI_SILU, half_t>(cTh,  c2w1t, c2_b1, nullptr, ct2h, Bc * CDc, Sc,  Sc, stream);
  gemm_launch<EPI_NONE, float >(ct2h, c2w2t, c2_b2, nullptr, cc,   Bc * CDc, CSc, Sc, stream);
  embnorm_kernel<<<Bc, 256, 0, stream>>>(cc, norm_emb_w, embv);
  gemm_launch<EPI_NONE, float >(embv, hd1wt, head1_b, nullptr, out + (size_t)BSc * Vc,
                                Bc, Sc * Vc, DEMBc, stream);

  (void)in_sizes; (void)n_in; (void)out_size; (void)ws_size;
}
